// DagCellTorch_19971597926403
// MI455X (gfx1250) — compile-verified, matrix operand order checked by reading, not run on hardware
//
#include <hip/hip_runtime.h>
#include <hip/hip_bf16.h>
#include <stdint.h>

// ---------------------------------------------------------------------------
// DAG recurrent cell for MI455X (gfx1250, wave32, WMMA).
//   T=128, B=512, H=ninp=256.  Per step: 18 GEMMs of 512x256x256 + gating.
// Strategy:
//   * pack all 18 weight matrices to bf16 once (they stay hot in L2)
//   * one workgroup per 64 batch rows; the full T-loop runs inside the kernel
//     (batch rows are independent -> no inter-workgroup sync ever needed)
//   * hidden state double-buffered in LDS as bf16 (WMMA A operand), final
//     per-step h also kept in f32 LDS for the norm-clip + coalesced output
//   * matrix core path: v_wmma_f32_16x16x32_bf16, f32 accumulate
//   * loops deliberately kept ROLLED (unroll 1) and col-tiles processed in
//     two half-passes so B fragments are consumed immediately -> no spills
// ---------------------------------------------------------------------------

typedef __attribute__((ext_vector_type(16))) __bf16        v16bf;
typedef __attribute__((ext_vector_type(8)))  float         v8f;
typedef __attribute__((ext_vector_type(4)))  float         f32x4;
typedef __attribute__((ext_vector_type(4)))  unsigned int  u32x4;
typedef __attribute__((ext_vector_type(4)))  unsigned short u16x4;

#define T_STEPS  128
#define BATCH    512
#define HID      256
#define MROWS    64          // batch rows per workgroup
#define NTHREADS 512         // 16 wave32s
#define NBLOCKS  (BATCH / MROWS)
#define MATELEMS (HID * HID) // 65536 elements per weight matrix

// LDS layout (dynamic shared memory), all 16B aligned offsets
#define SM_HB0   0                       // bf16 h, buffer 0   (64*256*2 = 32KB)
#define SM_HB1   (SM_HB0 + 32768)        // bf16 h, buffer 1
#define SM_XB    (SM_HB1 + 32768)        // bf16 x tile        (32KB)
#define SM_HF    (SM_XB  + 32768)        // f32  h (final per-step) (64KB)
#define SM_BXC   (SM_HF  + 65536)        // f32 bias xc (256)
#define SM_BXH   (SM_BXC + 1024)         // f32 bias xh (256)
#define SM_PART  (SM_BXH + 1024)         // f32 norm partials (512)
#define SM_SC    (SM_PART + 2048)        // f32 row scales (64)
#define SMEM_BYTES (SM_SC + 256)         // = 168192

// act codes: 0=tanh 1=relu 2=sigmoid 3=identity.
// ACTS[1..7] = {relu,tanh,sigmoid,identity,relu,tanh,identity}
// packed 3 bits per edge: (ACT_PACK >> (3*e)) & 7
#define ACT_PACK 792193  // 1|0<<3|2<<6|3<<9|1<<12|0<<15|3<<18

// ------------------------------- helpers -----------------------------------

__device__ __forceinline__ unsigned short f2bf(float f) {
  union { float f; unsigned int u; } c; c.f = f;
  unsigned int u = c.u;
  unsigned int r = u + 0x7FFFu + ((u >> 16) & 1u);   // round-to-nearest-even
  return (unsigned short)(r >> 16);
}
__device__ __forceinline__ float bf2f(unsigned short h) {
  union { unsigned int u; float f; } c; c.u = ((unsigned int)h) << 16;
  return c.f;
}
__device__ __forceinline__ float sigmoid_fast(float x) {
  return 1.0f / (1.0f + __expf(-x));
}
__device__ __forceinline__ float tanh_fast(float x) {
  // 1 - 2/(e^{2x}+1): saturates correctly at +/-1 for large |x|
  float e = __expf(2.0f * x);
  return 1.0f - 2.0f / (e + 1.0f);
}
__device__ __forceinline__ float act_apply(float v, int code) {
  if (code == 0) return tanh_fast(v);
  if (code == 1) return fmaxf(v, 0.0f);
  if (code == 2) return sigmoid_fast(v);
  return v;
}

union Frag16 { u32x4 q[2]; v16bf v; };

// A (16x32 bf16) and B (32x16 bf16) fragments share the same per-lane shape:
// 8 bf16 at element (base*256 + k0 + hi*8) and 8 bf16 at (+16 in K).
// For A: base = row m (LDS, row-major h). For B: base = n (global, row-major
// W[n][k] -> exactly W.T as the B matrix needs).
__device__ __forceinline__ v16bf load_frag16(const unsigned short* __restrict__ p,
                                             int base_row, int k0, int hi) {
  const int e0 = base_row * HID + k0 + hi * 8;
  Frag16 f;
  f.q[0] = *(const u32x4*)(p + e0);
  f.q[1] = *(const u32x4*)(p + e0 + 16);
  return f.v;
}

__device__ __forceinline__ v8f wmma_bf16(v16bf a, v16bf b, v8f c) {
  return __builtin_amdgcn_wmma_f32_16x16x32_bf16(
      /*neg_a=*/false, a, /*neg_b=*/false, b,
      /*c_mod=*/(short)0, c, /*reuse_a=*/false, /*reuse_b=*/false);
}

// ------------------------------ pack kernel --------------------------------

__global__ __launch_bounds__(256)
void pack_bf16_kernel(const float* __restrict__ src,
                      unsigned short* __restrict__ dst, int n4) {
  int i = blockIdx.x * 256 + threadIdx.x;
  if (i < n4) {
    f32x4 v = ((const f32x4*)src)[i];
    u16x4 o;
    o.x = f2bf(v.x); o.y = f2bf(v.y); o.z = f2bf(v.z); o.w = f2bf(v.w);
    ((u16x4*)dst)[i] = o;
  }
}

// ------------------------------ main kernel --------------------------------
// Wb layout (bf16, each MATELEMS): [0]=w_xc, [1]=w_xh, [2]=w_hc, [3]=w_hh,
//                                  [4..10]=Wc[0..6], [11..17]=Wh[0..6]

__global__ __launch_bounds__(NTHREADS)
void dag_cell_kernel(const float* __restrict__ x,
                     const float* __restrict__ h0,
                     const float* __restrict__ bxc,
                     const float* __restrict__ bxh,
                     const unsigned short* __restrict__ Wb,
                     float* __restrict__ out) {
  extern __shared__ __align__(16) char smem[];
  unsigned short* hb_cur = (unsigned short*)(smem + SM_HB0);
  unsigned short* hb_nxt = (unsigned short*)(smem + SM_HB1);
  unsigned short* XB   = (unsigned short*)(smem + SM_XB);
  float*          HF   = (float*)(smem + SM_HF);
  float*          SBXC = (float*)(smem + SM_BXC);
  float*          SBXH = (float*)(smem + SM_BXH);
  float*          PART = (float*)(smem + SM_PART);
  float*          SSC  = (float*)(smem + SM_SC);

  const int tid  = threadIdx.x;
  const int lane = tid & 31;
  const int wv   = tid >> 5;          // 0..15
  const int rt   = wv & 3;            // row-tile of this wave (16 rows)
  const int ctb  = (wv >> 2) * 4;     // first of 4 col-tiles for this wave
  const int hi   = lane >> 4;         // K-half selector (A/B frag layout)
  const int nlo  = lane & 15;
  const int arow = rt * 16 + nlo;     // A-fragment row for this lane
  const int row0 = blockIdx.x * MROWS;

  if (tid < HID) { SBXC[tid] = bxc[tid]; SBXH[tid] = bxh[tid]; }

  // initial hidden -> bf16 current buffer
  {
    const f32x4* h4 = (const f32x4*)(h0 + (size_t)row0 * HID);
    #pragma unroll
    for (int i = 0; i < 8; ++i) {
      int idx = i * NTHREADS + tid;        // 4096 float4s
      int r = idx >> 6, c4 = idx & 63;
      f32x4 v = h4[r * 64 + c4];
      u16x4 o; o.x = f2bf(v.x); o.y = f2bf(v.y); o.z = f2bf(v.z); o.w = f2bf(v.w);
      *(u16x4*)(hb_cur + r * HID + c4 * 4) = o;
    }
  }
  __syncthreads();

  #pragma unroll 1
  for (int t = 0; t < T_STEPS; ++t) {
    // ---- stage x_t tile as bf16 ----
    const f32x4* x4 = (const f32x4*)(x + (size_t)t * BATCH * HID + (size_t)row0 * HID);
    #pragma unroll
    for (int i = 0; i < 8; ++i) {
      int idx = i * NTHREADS + tid;
      int r = idx >> 6, c4 = idx & 63;
      f32x4 v = x4[r * 64 + c4];
      u16x4 o; o.x = f2bf(v.x); o.y = f2bf(v.y); o.z = f2bf(v.z); o.w = f2bf(v.w);
      *(u16x4*)(XB + r * HID + c4 * 4) = o;
    }
    __syncthreads();

    // ---- node 0: c0 = sig(x Wxc^T + bxc + h Whc^T);
    //              h  = c0*tanh(x Wxh^T + bxh + h Whh^T) + (1-c0)h
    {
      const unsigned short* Wxc = Wb + 0 * MATELEMS;
      const unsigned short* Wxh = Wb + 1 * MATELEMS;
      const unsigned short* Whc = Wb + 2 * MATELEMS;
      const unsigned short* Whh = Wb + 3 * MATELEMS;
      // two half-passes of 2 col-tiles each keeps live accumulators at 32 VGPRs
      #pragma unroll 1
      for (int jp = 0; jp < 4; jp += 2) {
        v8f accC[2] = {}; v8f accH[2] = {};
        #pragma unroll 1
        for (int kc = 0; kc < 8; ++kc) {
          const int k0 = kc * 32;
          v16bf Ah = load_frag16(hb_cur, arow, k0, hi);
          v16bf Ax = load_frag16(XB,     arow, k0, hi);
          #pragma unroll
          for (int j = 0; j < 2; ++j) {
            const int n = (ctb + jp + j) * 16 + nlo;
            accC[j] = wmma_bf16(Ah, load_frag16(Whc, n, k0, hi), accC[j]);
            accC[j] = wmma_bf16(Ax, load_frag16(Wxc, n, k0, hi), accC[j]);
            accH[j] = wmma_bf16(Ah, load_frag16(Whh, n, k0, hi), accH[j]);
            accH[j] = wmma_bf16(Ax, load_frag16(Wxh, n, k0, hi), accH[j]);
          }
        }
        #pragma unroll
        for (int j = 0; j < 2; ++j) {
          const int n = (ctb + jp + j) * 16 + nlo;
          const float bc = SBXC[n], bh = SBXH[n];
          #pragma unroll
          for (int r = 0; r < 8; ++r) {
            const int m = rt * 16 + (hi << 3) + r;   // C/D layout: M = r + 8*hi
            float c    = sigmoid_fast(accC[j][r] + bc);
            float f    = tanh_fast(accH[j][r] + bh);
            float hold = bf2f(hb_cur[m * HID + n]);
            float hn   = fmaf(c, f - hold, hold);    // c*f + (1-c)*hold
            hb_nxt[m * HID + n] = f2bf(hn);
          }
        }
      }
      __syncthreads();
      unsigned short* tmp = hb_cur; hb_cur = hb_nxt; hb_nxt = tmp;
    }

    // ---- edges 0..6 (nodes 1..7): c = sig(h Wc^T); h = c*act(h Wh^T) + (1-c)h
    #pragma unroll 1
    for (int e = 0; e < 7; ++e) {
      const int act = (ACT_PACK >> (3 * e)) & 7;
      const unsigned short* Wec = Wb + (4 + e)  * MATELEMS;
      const unsigned short* Weh = Wb + (11 + e) * MATELEMS;
      #pragma unroll 1
      for (int jp = 0; jp < 4; jp += 2) {
        v8f accC[2] = {}; v8f accH[2] = {};
        #pragma unroll 1
        for (int kc = 0; kc < 8; ++kc) {
          const int k0 = kc * 32;
          v16bf Ah = load_frag16(hb_cur, arow, k0, hi);
          #pragma unroll
          for (int j = 0; j < 2; ++j) {
            const int n = (ctb + jp + j) * 16 + nlo;
            accC[j] = wmma_bf16(Ah, load_frag16(Wec, n, k0, hi), accC[j]);
            accH[j] = wmma_bf16(Ah, load_frag16(Weh, n, k0, hi), accH[j]);
          }
        }
        #pragma unroll
        for (int j = 0; j < 2; ++j) {
          const int n = (ctb + jp + j) * 16 + nlo;
          #pragma unroll
          for (int r = 0; r < 8; ++r) {
            const int m = rt * 16 + (hi << 3) + r;
            float c    = sigmoid_fast(accC[j][r]);
            float f    = act_apply(accH[j][r], act);
            float hold = bf2f(hb_cur[m * HID + n]);
            float hn   = fmaf(c, f - hold, hold);
            hb_nxt[m * HID + n] = f2bf(hn);
            if (e == 6) HF[m * HID + n] = hn;   // last node: keep f32 copy
          }
        }
      }
      __syncthreads();
      unsigned short* tmp = hb_cur; hb_cur = hb_nxt; hb_nxt = tmp;
    }
    // 8 buffer flips per step -> hb_cur parity is step-invariant

    // ---- per-row norm clip (MAX_NORM = 25) + emit outputs ----
    {
      const int r = tid >> 3, sg = tid & 7;     // 64 rows x 8 segments
      float s = 0.0f;
      const float* rowp = HF + r * HID + sg * 32;
      #pragma unroll
      for (int i = 0; i < 32; ++i) { float v = rowp[i]; s = fmaf(v, v, s); }
      PART[tid] = s;
    }
    __syncthreads();
    if (tid < MROWS) {
      float s = 0.0f;
      #pragma unroll
      for (int i = 0; i < 8; ++i) s += PART[tid * 8 + i];
      float nrm = sqrtf(s);
      SSC[tid] = (nrm > 25.0f) ? (25.0f / nrm) : 1.0f;
    }
    __syncthreads();
    {
      float* outp = out + (size_t)t * BATCH * HID + (size_t)row0 * HID;
      #pragma unroll
      for (int i = 0; i < 8; ++i) {
        int idx = i * NTHREADS + tid;
        int r = idx >> 6, c4 = idx & 63;
        f32x4 v = *(f32x4*)(HF + r * HID + c4 * 4);
        float sc = SSC[r];
        v.x *= sc; v.y *= sc; v.z *= sc; v.w *= sc;
        *(f32x4*)(outp + r * HID + c4 * 4) = v;      // coalesced b128 store
        *(f32x4*)(HF + r * HID + c4 * 4) = v;        // scaled f32 state
        u16x4 o; o.x = f2bf(v.x); o.y = f2bf(v.y); o.z = f2bf(v.z); o.w = f2bf(v.w);
        *(u16x4*)(hb_cur + r * HID + c4 * 4) = o;    // scaled bf16 state
      }
    }
    __syncthreads();
  }

  // ---- hidden_final (second output, appended after the T*B*H outputs) ----
  {
    float* hfo = out + (size_t)T_STEPS * BATCH * HID + (size_t)row0 * HID;
    #pragma unroll
    for (int i = 0; i < 8; ++i) {
      int idx = i * NTHREADS + tid;
      int r = idx >> 6, c4 = idx & 63;
      *(f32x4*)(hfo + r * HID + c4 * 4) = *(f32x4*)(HF + r * HID + c4 * 4);
    }
  }
}

// ------------------------------- launcher ----------------------------------

extern "C" void kernel_launch(void* const* d_in, const int* in_sizes, int n_in,
                              void* d_out, int out_size, void* d_ws, size_t ws_size,
                              hipStream_t stream) {
  (void)in_sizes; (void)n_in; (void)out_size; (void)ws_size;
  const float* x   = (const float*)d_in[0];
  const float* h0  = (const float*)d_in[1];
  const float* wxc = (const float*)d_in[2];
  const float* bxc = (const float*)d_in[3];
  const float* wxh = (const float*)d_in[4];
  const float* bxh = (const float*)d_in[5];
  const float* whc = (const float*)d_in[6];
  const float* whh = (const float*)d_in[7];
  const float* Wc  = (const float*)d_in[8];
  const float* Wh  = (const float*)d_in[9];

  unsigned short* Wb = (unsigned short*)d_ws;   // 18 * 65536 bf16 = 2.36 MB

  auto pack = [&](const float* s, unsigned short* d, int elems) {
    int n4 = elems / 4;
    pack_bf16_kernel<<<(n4 + 255) / 256, 256, 0, stream>>>(s, d, n4);
  };
  pack(wxc, Wb + 0  * MATELEMS, MATELEMS);
  pack(wxh, Wb + 1  * MATELEMS, MATELEMS);
  pack(whc, Wb + 2  * MATELEMS, MATELEMS);
  pack(whh, Wb + 3  * MATELEMS, MATELEMS);
  pack(Wc,  Wb + 4  * MATELEMS, 7 * MATELEMS);
  pack(Wh,  Wb + 11 * MATELEMS, 7 * MATELEMS);

  // 168 KB dynamic LDS (gfx1250 WGP has 320 KB); opt in above the default cap.
  (void)hipFuncSetAttribute((const void*)dag_cell_kernel,
                            hipFuncAttributeMaxDynamicSharedMemorySize,
                            SMEM_BYTES);

  dag_cell_kernel<<<NBLOCKS, NTHREADS, SMEM_BYTES, stream>>>(
      x, h0, bxc, bxh, Wb, (float*)d_out);
}
